// MultiStepLSTMDecoder_79156247265951
// MI455X (gfx1250) — matrix operand order repaired and checked by reading, not compiled, not run on hardware
//
#include <hip/hip_runtime.h>
#include <hip/hip_bf16.h>
#include <math.h>

// ---------------------------------------------------------------------------
// MultiStepLSTMDecoder on gfx1250 (MI455X), wave32 + v_wmma_f32_16x16x32_bf16
// B=64, S=512, H=512, L=2, NSTEPS=24
// ---------------------------------------------------------------------------

typedef __bf16 bf16_t;
typedef __attribute__((ext_vector_type(16))) __bf16 v16bf;
typedef __attribute__((ext_vector_type(8)))  float  v8f;

#define Bn 64
#define Sn 512
#define Hn 512
#define NSTEP 24
#define LN_EPS 1e-5f

// --------------------------- utility kernels -------------------------------

__global__ void k_f32_to_bf16(const float* __restrict__ src, bf16_t* __restrict__ dst, int n) {
    int i = blockIdx.x * blockDim.x + threadIdx.x;
    if (i < n) dst[i] = (bf16_t)src[i];
}

__global__ void k_copy_f32(const float* __restrict__ src, float* __restrict__ dst, int n) {
    int i = blockIdx.x * blockDim.x + threadIdx.x;
    if (i < n) dst[i] = src[i];
}

__global__ void k_bias_combine(const float* __restrict__ a, const float* __restrict__ b,
                               float* __restrict__ out, int n) {
    int i = blockIdx.x * blockDim.x + threadIdx.x;
    if (i < n) out[i] = a[i] + b[i];
}

// --------------------------- WMMA GEMM kernel ------------------------------
// D[M,N] = A1[M,K1] * B1t[N,K1]^T + A2[M,K2] * B2t[N,K2]^T + bias[N]
// A row-major bf16; B stored as [N][K] (i.e. original torch weight [out,in]),
// so logical-B column n == stored row n, contiguous in K.
// Block = (32,4): 4 waves, each computes one 16x16 f32 tile. M % 64 == 0.
__global__ void k_wmma_gemm(const bf16_t* __restrict__ A1, const bf16_t* __restrict__ B1, int K1,
                            const bf16_t* __restrict__ A2, const bf16_t* __restrict__ B2, int K2,
                            const float* __restrict__ bias, float* __restrict__ C, int N) {
    const int lane  = threadIdx.x;          // 0..31
    const int tileM = blockIdx.y * 4 + threadIdx.y;
    const int tileN = blockIdx.x;
    const int row   = lane & 15;
    const int half  = lane >> 4;
    const int m_base = tileM * 16;
    const int n      = tileN * 16 + row;    // output column handled by this lane

    v8f acc = {};

    // ---- first A/B pair ----
    for (int kc = 0; kc < K1; kc += 32) {
        v16bf a, b;
        const bf16_t* Ar = A1 + (size_t)(m_base + row) * K1 + kc;
#pragma unroll
        for (int vv = 0; vv < 8; ++vv) {
            int k0 = (vv < 4 ? 2 * vv : 16 + 2 * (vv - 4)) + half * 8;
            a[2 * vv]     = Ar[k0];
            a[2 * vv + 1] = Ar[k0 + 1];
        }
        const bf16_t* Bc = B1 + (size_t)n * K1 + kc + half * 8;
#pragma unroll
        for (int e = 0; e < 16; ++e) b[e] = Bc[e < 8 ? e : e + 8];
        acc = __builtin_amdgcn_wmma_f32_16x16x32_bf16(false, a, false, b,
                                                      (short)0, acc, false, false);
    }

    // ---- optional second A/B pair (fused h @ Whh^T for LSTM gates) ----
    for (int kc = 0; kc < K2; kc += 32) {
        v16bf a, b;
        const bf16_t* Ar = A2 + (size_t)(m_base + row) * K2 + kc;
#pragma unroll
        for (int vv = 0; vv < 8; ++vv) {
            int k0 = (vv < 4 ? 2 * vv : 16 + 2 * (vv - 4)) + half * 8;
            a[2 * vv]     = Ar[k0];
            a[2 * vv + 1] = Ar[k0 + 1];
        }
        const bf16_t* Bc = B2 + (size_t)n * K2 + kc + half * 8;
#pragma unroll
        for (int e = 0; e < 16; ++e) b[e] = Bc[e < 8 ? e : e + 8];
        acc = __builtin_amdgcn_wmma_f32_16x16x32_bf16(false, a, false, b,
                                                      (short)0, acc, false, false);
    }

    const float bv = bias ? bias[n] : 0.0f;
#pragma unroll
    for (int vv = 0; vv < 8; ++vv) {
        int m = m_base + vv + 8 * half;     // C/D layout: VGPR v -> M = v + 8*half
        C[(size_t)m * N + n] = acc[vv] + bv;
    }
}

// --------------------------- per-step kernels ------------------------------

// Stage old h-state as bf16 A operands; also fill xcat[:,512:1024] = h_prev.
__global__ void k_stage(const float* __restrict__ h_state,   // [2,B,H]
                        bf16_t* __restrict__ hbf0, bf16_t* __restrict__ hbf1,
                        bf16_t* __restrict__ xcat) {
    int i = blockIdx.x * blockDim.x + threadIdx.x;            // < B*H
    if (i >= Bn * Hn) return;
    hbf0[i] = (bf16_t)h_state[i];
    float h1 = h_state[Bn * Hn + i];
    hbf1[i] = (bf16_t)h1;
    int b = i >> 9, hh = i & 511;
    xcat[b * 1024 + 512 + hh] = (bf16_t)h1;
}

// scores[b,s] = sum_h v[h] * tanh(keys[b,s,h] + q[b,h]); one wave per (b,s).
__global__ void k_attn_scores(const float* __restrict__ keys, const float* __restrict__ q,
                              const float* __restrict__ v, float* __restrict__ scores) {
    int rowid = blockIdx.x * blockDim.y + threadIdx.y;        // 0..B*S-1
    int b = rowid >> 9;
    const float* krow = keys + (size_t)rowid * Hn;
    const float* qrow = q + b * Hn;
    float acc = 0.0f;
    for (int h = threadIdx.x; h < Hn; h += 32) {
        __builtin_prefetch(krow + h + 128, 0, 1);             // global_prefetch_b8
        acc += v[h] * tanhf(krow[h] + qrow[h]);
    }
#pragma unroll
    for (int off = 16; off > 0; off >>= 1) acc += __shfl_xor(acc, off, 32);
    if (threadIdx.x == 0) scores[rowid] = acc;
}

// in-place softmax over S per batch row; grid=B, block=S.
__global__ void k_softmax(float* __restrict__ scores) {
    __shared__ float red[Sn];
    int b = blockIdx.x, s = threadIdx.x;
    float x = scores[b * Sn + s];
    red[s] = x; __syncthreads();
    for (int st = Sn / 2; st > 0; st >>= 1) {
        if (s < st) red[s] = fmaxf(red[s], red[s + st]);
        __syncthreads();
    }
    float m = red[0]; __syncthreads();
    float e = expf(x - m);
    red[s] = e; __syncthreads();
    for (int st = Sn / 2; st > 0; st >>= 1) {
        if (s < st) red[s] += red[s + st];
        __syncthreads();
    }
    scores[b * Sn + s] = e / red[0];
}

// context[b,h] = sum_s attn[b,s]*enc[b,s,h]; writes bf16 into xcat[:,0:512].
__global__ void k_context(const float* __restrict__ attn, const float* __restrict__ enc,
                          bf16_t* __restrict__ xcat) {
    __shared__ float aw[Sn];
    int b = blockIdx.x, t = threadIdx.x;                      // block = 256
    aw[t] = attn[b * Sn + t];
    aw[t + 256] = attn[b * Sn + t + 256];
    __syncthreads();
    const float* e = enc + (size_t)b * Sn * Hn;
    float acc0 = 0.0f, acc1 = 0.0f;
    for (int s = 0; s < Sn; ++s) {
        float a = aw[s];
        acc0 += a * e[(size_t)s * Hn + t];
        acc1 += a * e[(size_t)s * Hn + t + 256];
    }
    xcat[b * 1024 + t]       = (bf16_t)acc0;
    xcat[b * 1024 + t + 256] = (bf16_t)acc1;
}

// LSTM cell nonlinearity (gate order i,f,g,o); updates h/c in place, stages bf16 h.
__global__ void k_lstm_update(const float* __restrict__ gates,
                              float* __restrict__ h_l, float* __restrict__ c_l,
                              bf16_t* __restrict__ hbf_out) {
    int i = blockIdx.x * blockDim.x + threadIdx.x;            // < B*H
    if (i >= Bn * Hn) return;
    int b = i >> 9, hh = i & 511;
    const float* g = gates + (size_t)b * 2048;
    float ig = g[hh], fg = g[hh + 512], gg = g[hh + 1024], og = g[hh + 1536];
    float si = 1.0f / (1.0f + expf(-ig));
    float sf = 1.0f / (1.0f + expf(-fg));
    float so = 1.0f / (1.0f + expf(-og));
    float cn = sf * c_l[i] + si * tanhf(gg);
    float hn = so * tanhf(cn);
    c_l[i] = cn;
    h_l[i] = hn;
    hbf_out[i] = (bf16_t)hn;
}

// LayerNorm over H; writes bf16 (GEMM A operand). grid=B, block=256, 2 elems/thr.
__global__ void k_layernorm(const float* __restrict__ x, const float* __restrict__ g,
                            const float* __restrict__ bta, bf16_t* __restrict__ xn) {
    __shared__ float ssum[256], ssq[256];
    int b = blockIdx.x, t = threadIdx.x;
    float x0 = x[b * Hn + t], x1 = x[b * Hn + t + 256];
    ssum[t] = x0 + x1;
    ssq[t]  = x0 * x0 + x1 * x1;
    __syncthreads();
    for (int st = 128; st > 0; st >>= 1) {
        if (t < st) { ssum[t] += ssum[t + st]; ssq[t] += ssq[t + st]; }
        __syncthreads();
    }
    float mu  = ssum[0] * (1.0f / Hn);
    float var = ssq[0] * (1.0f / Hn) - mu * mu;
    float inv = rsqrtf(var + LN_EPS);
    xn[b * Hn + t]       = (bf16_t)((x0 - mu) * inv * g[t]       + bta[t]);
    xn[b * Hn + t + 256] = (bf16_t)((x1 - mu) * inv * g[t + 256] + bta[t + 256]);
}

// preds[b, step] = b2 + sum_j w2[j]*relu(z1[b,j]); grid=B, block=256.
__global__ void k_head_out(const float* __restrict__ z1, const float* __restrict__ w2,
                           const float* __restrict__ b2, float* __restrict__ preds, int step) {
    __shared__ float red[256];
    int b = blockIdx.x, t = threadIdx.x;
    red[t] = fmaxf(z1[b * 256 + t], 0.0f) * w2[t];
    __syncthreads();
    for (int st = 128; st > 0; st >>= 1) {
        if (t < st) red[t] += red[t + st];
        __syncthreads();
    }
    if (t == 0) preds[b * NSTEP + step] = red[0] + b2[0];
}

// --------------------------- host orchestration ----------------------------

static inline void launch_gemm(hipStream_t s,
                               const bf16_t* A1, const bf16_t* B1, int K1,
                               const bf16_t* A2, const bf16_t* B2, int K2,
                               const float* bias, float* C, int M, int N) {
    dim3 grid(N / 16, M / 64), block(32, 4);
    hipLaunchKernelGGL(k_wmma_gemm, grid, block, 0, s, A1, B1, K1, A2, B2, K2, bias, C, N);
}

extern "C" void kernel_launch(void* const* d_in, const int* in_sizes, int n_in,
                              void* d_out, int out_size, void* d_ws, size_t ws_size,
                              hipStream_t stream) {
    // ---- inputs (setup_inputs order) ----
    const float* enc   = (const float*)d_in[0];   // [B,S,H]
    const float* h0    = (const float*)d_in[1];   // [2,B,H]
    const float* c0    = (const float*)d_in[2];
    const float* Wq    = (const float*)d_in[3];   // [H,H]
    const float* bq    = (const float*)d_in[4];
    const float* Wk    = (const float*)d_in[5];
    const float* bk    = (const float*)d_in[6];
    const float* vvec  = (const float*)d_in[7];
    const float* wih0  = (const float*)d_in[8];   // [2048,1024]
    const float* whh0  = (const float*)d_in[9];   // [2048,512]
    const float* bih0  = (const float*)d_in[10];
    const float* bhh0  = (const float*)d_in[11];
    const float* wih1  = (const float*)d_in[12];  // [2048,512]
    const float* whh1  = (const float*)d_in[13];
    const float* bih1  = (const float*)d_in[14];
    const float* bhh1  = (const float*)d_in[15];
    const float* ln_g  = (const float*)d_in[16];
    const float* ln_b  = (const float*)d_in[17];
    const float* w1    = (const float*)d_in[18];  // [256,512]
    const float* b1    = (const float*)d_in[19];
    const float* w2    = (const float*)d_in[20];  // [256]
    const float* b2    = (const float*)d_in[21];

    // ---- outputs: preds[1536] | h[2*B*H] | c[2*B*H]; h/c states live in d_out ----
    float* preds   = (float*)d_out;
    float* h_state = preds + Bn * NSTEP;          // [2,B,H]
    float* c_state = h_state + 2 * Bn * Hn;

    // ---- workspace carve-out ----
    size_t off = 0;
    char* base = (char*)d_ws;
    auto carve = [&](size_t bytes) -> void* {
        void* p = base + off;
        off = (off + bytes + 255) & ~(size_t)255;
        return p;
    };
    const size_t BSH = (size_t)Bn * Sn * Hn;
    float*  keys     = (float*)carve(BSH * 4);
    bf16_t* enc_bf   = (bf16_t*)carve(BSH * 2);
    bf16_t* wq_bf    = (bf16_t*)carve((size_t)Hn * Hn * 2);
    bf16_t* wk_bf    = (bf16_t*)carve((size_t)Hn * Hn * 2);
    bf16_t* wih0_bf  = (bf16_t*)carve((size_t)2048 * 1024 * 2);
    bf16_t* whh0_bf  = (bf16_t*)carve((size_t)2048 * 512 * 2);
    bf16_t* wih1_bf  = (bf16_t*)carve((size_t)2048 * 512 * 2);
    bf16_t* whh1_bf  = (bf16_t*)carve((size_t)2048 * 512 * 2);
    bf16_t* w1_bf    = (bf16_t*)carve((size_t)256 * 512 * 2);
    float*  bsum0    = (float*)carve(2048 * 4);
    float*  bsum1    = (float*)carve(2048 * 4);
    float*  qbuf     = (float*)carve((size_t)Bn * Hn * 4);
    float*  scores   = (float*)carve((size_t)Bn * Sn * 4);
    bf16_t* xcat_bf  = (bf16_t*)carve((size_t)Bn * 1024 * 2);
    bf16_t* hbf0     = (bf16_t*)carve((size_t)Bn * Hn * 2);
    bf16_t* hbf1     = (bf16_t*)carve((size_t)Bn * Hn * 2);
    bf16_t* h0new_bf = (bf16_t*)carve((size_t)Bn * Hn * 2);
    float*  gates    = (float*)carve((size_t)Bn * 2048 * 4);
    bf16_t* xn_bf    = (bf16_t*)carve((size_t)Bn * Hn * 2);
    float*  z1       = (float*)carve((size_t)Bn * 256 * 4);

    auto cvt = [&](const float* s, bf16_t* d, size_t n) {
        hipLaunchKernelGGL(k_f32_to_bf16, dim3((n + 255) / 256), dim3(256), 0, stream,
                           s, d, (int)n);
    };

    // ---- one-time setup: weight/bias staging, keys GEMM, state init ----
    cvt(enc, enc_bf, BSH);
    cvt(Wq, wq_bf, (size_t)Hn * Hn);
    cvt(Wk, wk_bf, (size_t)Hn * Hn);
    cvt(wih0, wih0_bf, (size_t)2048 * 1024);
    cvt(whh0, whh0_bf, (size_t)2048 * 512);
    cvt(wih1, wih1_bf, (size_t)2048 * 512);
    cvt(whh1, whh1_bf, (size_t)2048 * 512);
    cvt(w1, w1_bf, (size_t)256 * 512);
    hipLaunchKernelGGL(k_bias_combine, dim3(8), dim3(256), 0, stream, bih0, bhh0, bsum0, 2048);
    hipLaunchKernelGGL(k_bias_combine, dim3(8), dim3(256), 0, stream, bih1, bhh1, bsum1, 2048);
    hipLaunchKernelGGL(k_copy_f32, dim3((2 * Bn * Hn + 255) / 256), dim3(256), 0, stream,
                       h0, h_state, 2 * Bn * Hn);
    hipLaunchKernelGGL(k_copy_f32, dim3((2 * Bn * Hn + 255) / 256), dim3(256), 0, stream,
                       c0, c_state, 2 * Bn * Hn);

    // keys = enc @ Wk^T + bk   (M=32768, N=512, K=512) — the big WMMA GEMM
    launch_gemm(stream, enc_bf, wk_bf, Hn, nullptr, nullptr, 0, bk, keys, Bn * Sn, Hn);

    const int bhBlocks = (Bn * Hn + 255) / 256;

    // ---- 24 sequential decode steps ----
    for (int t = 0; t < NSTEP; ++t) {
        // stage old h as bf16 (also xcat[:,512:] = h_prev)
        hipLaunchKernelGGL(k_stage, dim3(bhBlocks), dim3(256), 0, stream,
                           h_state, hbf0, hbf1, xcat_bf);
        // q = h_prev @ Wq^T + bq
        launch_gemm(stream, hbf1, wq_bf, Hn, nullptr, nullptr, 0, bq, qbuf, Bn, Hn);
        // attention scores + softmax + context
        hipLaunchKernelGGL(k_attn_scores, dim3(Bn * Sn / 8), dim3(32, 8), 0, stream,
                           keys, qbuf, vvec, scores);
        hipLaunchKernelGGL(k_softmax, dim3(Bn), dim3(Sn), 0, stream, scores);
        hipLaunchKernelGGL(k_context, dim3(Bn), dim3(256), 0, stream, scores, enc, xcat_bf);
        // LSTM layer 0: gates = xcat @ Wih0^T + h0_old @ Whh0^T + (b_ih0+b_hh0)
        launch_gemm(stream, xcat_bf, wih0_bf, 1024, hbf0, whh0_bf, 512,
                    bsum0, gates, Bn, 2048);
        hipLaunchKernelGGL(k_lstm_update, dim3(bhBlocks), dim3(256), 0, stream,
                           gates, h_state, c_state, h0new_bf);
        // LSTM layer 1: gates = h0_new @ Wih1^T + h1_old @ Whh1^T + (b_ih1+b_hh1)
        launch_gemm(stream, h0new_bf, wih1_bf, 512, hbf1, whh1_bf, 512,
                    bsum1, gates, Bn, 2048);
        hipLaunchKernelGGL(k_lstm_update, dim3(bhBlocks), dim3(256), 0, stream,
                           gates, h_state + Bn * Hn, c_state + Bn * Hn, hbf1 /*scratch*/);
        // head: LayerNorm -> Linear(512->256)+bias -> ReLU -> Linear(256->1)
        hipLaunchKernelGGL(k_layernorm, dim3(Bn), dim3(256), 0, stream,
                           h_state + Bn * Hn, ln_g, ln_b, xn_bf);
        launch_gemm(stream, xn_bf, w1_bf, Hn, nullptr, nullptr, 0, b1, z1, Bn, 256);
        hipLaunchKernelGGL(k_head_out, dim3(Bn), dim3(256), 0, stream, z1, w2, b2, preds, t);
    }
}
